// NodeAttention_46514495816185
// MI455X (gfx1250) — matrix operand order repaired and checked by reading, not compiled, not run on hardware
//
#include <hip/hip_runtime.h>

// ---------------------------------------------------------------------------
// Problem constants (match reference)
// ---------------------------------------------------------------------------
#define NN 131072
#define DD 128
#define GG 512
#define LMAX 448
#define EPSV 1e-5f
#define PADG (LMAX * DD)     // elements per graph in padded interleaved V

typedef __attribute__((ext_vector_type(16))) __bf16 v16bf;
typedef __attribute__((ext_vector_type(8)))  float  v8f;

__device__ __forceinline__ v8f wmma_bf16(v16bf a, v16bf b, v8f c) {
    // D = A(16x32 bf16) x B(32x16 bf16) + C(16x16 f32)
    return __builtin_amdgcn_wmma_f32_16x16x32_bf16(
        false, a, false, b, (short)0, c, false, false);
}

// A-matrix (16-bit, 16x32) per-lane K index for VGPR pair v (0..7), half h:
//   K = 16h + 2v (contiguous order, matches B fills)
__device__ __forceinline__ int a_kpat(int v, int h) {
    return 16 * h + 2 * v;
}

// Assemble a 16xbf16 WMMA operand from two 16B-aligned memory runs.
__device__ __forceinline__ v16bf ld_2x128(const void* p0, const void* p1) {
    v16bf r;
    ((uint4*)&r)[0] = *(const uint4*)p0;
    ((uint4*)&r)[1] = *(const uint4*)p1;
    return r;
}

// --- CDNA5 async global->LDS copy (ASYNCcnt-tracked DMA, ISA 15.18.3) ------
__device__ __forceinline__ unsigned lds_addr(const void* p) {
    return (unsigned)(size_t)p;   // low 32 bits of shared aperture = LDS offset
}
__device__ __forceinline__ void async_b128(unsigned lds_off, const void* gptr) {
    asm volatile("global_load_async_to_lds_b128 %0, %1, off"
                 :: "v"(lds_off), "v"(gptr) : "memory");
}
__device__ __forceinline__ void wait_async0() {
    asm volatile("s_wait_asynccnt 0" ::: "memory");
}

// ---------------------------------------------------------------------------
// Segment bookkeeping: counts + exclusive-scan starts
// ---------------------------------------------------------------------------
__global__ void zero_kernel(int* counts) {
    int i = blockIdx.x * blockDim.x + threadIdx.x;
    if (i < GG) counts[i] = 0;
}

__global__ void hist_kernel(const int* __restrict__ batch, int* __restrict__ counts) {
    int i = blockIdx.x * blockDim.x + threadIdx.x;
    if (i < NN) atomicAdd(&counts[batch[i]], 1);
}

__global__ void scan_kernel(const int* __restrict__ counts, int* __restrict__ starts) {
    __shared__ int tmp[GG];
    int t = threadIdx.x;
    int v = counts[t];
    tmp[t] = v;
    __syncthreads();
    for (int off = 1; off < GG; off <<= 1) {
        int a = (t >= off) ? tmp[t - off] : 0;
        __syncthreads();
        tmp[t] += a;
        __syncthreads();
    }
    starts[t] = tmp[t] - v;   // exclusive prefix
}

// ---------------------------------------------------------------------------
// Projections: K/Q/V = x @ W^T + b.  One wave = 32 rows x 128 cols, 64 WMMAs.
// K,Q -> row-major bf16 [node][dim].
// V   -> per-graph padded, 8-row interleaved bf16:
//        Vp8[g*PADG + (pos>>3)*1024 + dim*8 + (pos&7)]
// so the attention Z^T A-operand fill is two b128 loads per lane.
// ---------------------------------------------------------------------------
__global__ __launch_bounds__(32)
void proj_kernel(const float* __restrict__ x,
                 const float* __restrict__ Wk, const float* __restrict__ bk,
                 const float* __restrict__ Wq, const float* __restrict__ bq,
                 const float* __restrict__ Wv, const float* __restrict__ bv,
                 const int* __restrict__ batch, const int* __restrict__ starts,
                 __bf16* __restrict__ Kb, __bf16* __restrict__ Qb,
                 __bf16* __restrict__ Vp8)
{
    const int lane = threadIdx.x;
    const int nn = lane & 15, h = lane >> 4;
    const int r0 = blockIdx.x * 32;
    const bool isV = (blockIdx.y == 2);

    const float* W; const float* bias;
    if (blockIdx.y == 0)      { W = Wk; bias = bk; }
    else if (blockIdx.y == 1) { W = Wq; bias = bq; }
    else                      { W = Wv; bias = bv; }

    // Hoisted segment lookups for the V interleaved store path
    size_t vdst0[8], vdst1[8];
    if (isV) {
        #pragma unroll
        for (int v = 0; v < 8; ++v) {
            #pragma unroll
            for (int rt = 0; rt < 2; ++rt) {
                int row = r0 + rt * 16 + v + 8 * h;
                int gg = batch[row];
                int pos = row - starts[gg];
                size_t base = (size_t)gg * PADG + (size_t)(pos >> 3) * 1024 + (pos & 7);
                if (rt) vdst1[v] = base; else vdst0[v] = base;
            }
        }
    }

    // Two persistent A tiles (rows of x), 4 chunks of K=32 over d_in=128
    v16bf a0[4], a1[4];
    {
        const float* s0p = x + (size_t)(r0 + nn) * DD;
        const float* s1p = x + (size_t)(r0 + 16 + nn) * DD;
        #pragma unroll
        for (int c = 0; c < 4; ++c)
            #pragma unroll
            for (int v = 0; v < 8; ++v) {
                int k = c * 32 + a_kpat(v, h);
                a0[c][2 * v]     = (__bf16)s0p[k];
                a0[c][2 * v + 1] = (__bf16)s0p[k + 1];
                a1[c][2 * v]     = (__bf16)s1p[k];
                a1[c][2 * v + 1] = (__bf16)s1p[k + 1];
            }
    }

    #pragma unroll
    for (int t = 0; t < 8; ++t) {
        v8f acc0 = {}, acc1 = {};
        #pragma unroll
        for (int c = 0; c < 4; ++c) {
            const float* wsrc = W + (size_t)(t * 16 + nn) * DD + c * 32 + 16 * h;
            v16bf b;
            #pragma unroll
            for (int v = 0; v < 8; ++v) {
                b[2 * v]     = (__bf16)wsrc[2 * v];
                b[2 * v + 1] = (__bf16)wsrc[2 * v + 1];
            }
            acc0 = wmma_bf16(a0[c], b, acc0);
            acc1 = wmma_bf16(a1[c], b, acc1);
        }
        const float bb = bias[t * 16 + nn];
        const int dim = t * 16 + nn;
        if (isV) {                   // V: padded interleaved per-graph layout
            #pragma unroll
            for (int v = 0; v < 8; ++v) {
                Vp8[vdst0[v] + (size_t)dim * 8] = (__bf16)(acc0[v] + bb);
                Vp8[vdst1[v] + (size_t)dim * 8] = (__bf16)(acc1[v] + bb);
            }
        } else {
            __bf16* o = (blockIdx.y == 0) ? Kb : Qb;
            #pragma unroll
            for (int v = 0; v < 8; ++v) {
                o[(size_t)(r0 + v + 8 * h) * DD + dim]      = (__bf16)(acc0[v] + bb);
                o[(size_t)(r0 + 16 + v + 8 * h) * DD + dim] = (__bf16)(acc1[v] + bb);
            }
        }
    }
}

// ---------------------------------------------------------------------------
// Flash attention + residual + LayerNorm.
// Workgroup = 4 waves = 4 consecutive 16-row i-tiles of one graph, sharing a
// double-buffered async-DMA Q stream in LDS.  Z is computed transposed
// (V^T as WMMA-A from the interleaved global layout, P as WMMA-B from LDS).
// ---------------------------------------------------------------------------
#define PCH  40                    // bf16 prob row stride (16B-aligned rows)
#define ZSTR 132                   // f32 z row stride (16B-aligned rows)
#define SQ_BYTES   (2 * 32 * DD * 2)          // 16384: Q double buffer
#define SP_BYTES   (4 * 16 * PCH * 2)         // 5120:  per-wave P chunks
#define SZ_BYTES   (4 * 16 * ZSTR * 4)        // 33792: per-wave z tiles (aliases)
#define SMEM_BYTES (SZ_BYTES > (SQ_BYTES + SP_BYTES) ? SZ_BYTES : (SQ_BYTES + SP_BYTES))

__global__ __launch_bounds__(128)
void attn_kernel(const __bf16* __restrict__ Kb, const __bf16* __restrict__ Qb,
                 const __bf16* __restrict__ Vp8, const float* __restrict__ x,
                 const float* __restrict__ gamma, const float* __restrict__ beta,
                 const int* __restrict__ counts, const int* __restrict__ starts,
                 float* __restrict__ out)
{
    __shared__ char  smem[SMEM_BYTES];
    __shared__ float sSum[4][16];

    const int g = blockIdx.y;
    const int n = counts[g];
    if (blockIdx.x * 64 >= n) return;          // whole-block uniform exit
    const int s0 = starts[g];

    const int tid  = threadIdx.x;
    const int wid  = tid >> 5;
    const int lane = tid & 31;
    const int nn = lane & 15, h = lane >> 4;
    const int i0 = (blockIdx.x * 4 + wid) * 16;   // this wave's i-tile
    const float scale = 0.08838834764831845f;     // 1/sqrt(128)

    __bf16* sQ  = (__bf16*)smem;                           // [2][32*DD]
    __bf16* sPw = (__bf16*)(smem + SQ_BYTES) + wid * 16 * PCH;
    const __bf16* Vg = Vp8 + (size_t)g * PADG;

    // ---- persistent A operand: 16 K-rows, 4 chunks over d=128 -------------
    v16bf aK[4];
    {
        int row = s0 + min(i0 + nn, n - 1);
        const __bf16* src = Kb + (size_t)row * DD;
        #pragma unroll
        for (int c = 0; c < 4; ++c)
            #pragma unroll
            for (int v = 0; v < 8; ++v) {
                int k = c * 32 + a_kpat(v, h);
                aK[c][2 * v]     = src[k];
                aK[c][2 * v + 1] = src[k + 1];
            }
    }

    v8f zc[8];                      // Z^T: rows = dims (v+8h), cols = i (nn)
    #pragma unroll
    for (int t = 0; t < 8; ++t) zc[t] = (v8f){};
    float m8[8], l8[8];             // per score-row (i = v+8h) running max/sum
    #pragma unroll
    for (int v = 0; v < 8; ++v) { m8[v] = -1e30f; l8[v] = 0.f; }

    const int njc = (n + 31) >> 5;

    auto stage = [&](int jc, int buf) {   // all 128 threads: 32 Q rows -> LDS
        int rowLocal = tid >> 2;          // 0..31
        int part = tid & 3;               // 64B slice of the 256B row
        int row = s0 + min(jc * 32 + rowLocal, n - 1);
        const __bf16* qg = Qb + (size_t)row * DD + part * 32;
        unsigned qdst = lds_addr(sQ + buf * (32 * DD) + rowLocal * DD + part * 32);
        #pragma unroll
        for (int q = 0; q < 4; ++q) async_b128(qdst + q * 16, qg + q * 8);
    };

    stage(0, 0);
    wait_async0();
    __syncthreads();

    for (int jc = 0; jc < njc; ++jc) {
        const int cur = jc & 1;
        if (jc + 1 < njc) stage(jc + 1, cur ^ 1);   // DMA next chunk during compute
        const __bf16* Qs = sQ + cur * (32 * DD);

        // ---- two 16x16 score tiles: S[i,j], i=v+8h rows, j=nn cols --------
        v8f st[2];
        #pragma unroll
        for (int jt = 0; jt < 2; ++jt) {
            v8f c = {};
            #pragma unroll
            for (int ch = 0; ch < 4; ++ch) {
                const __bf16* q2 = Qs + (jt * 16 + nn) * DD + ch * 32 + 16 * h;
                v16bf b = ld_2x128(q2, q2 + 8);
                c = wmma_bf16(aK[ch], b, c);
            }
            st[jt] = c;
        }

        // ---- scale + column masking (column index depends only on lane) ---
        const bool ok0 = (jc * 32 + nn) < n;
        const bool ok1 = (jc * 32 + 16 + nn) < n;
        #pragma unroll
        for (int v = 0; v < 8; ++v) {
            st[0][v] = ok0 ? st[0][v] * scale : -1e30f;
            st[1][v] = ok1 ? st[1][v] * scale : -1e30f;
        }

        // ---- online softmax (per row v+8h; reduce across 16-lane half) ----
        float corr[8];
        #pragma unroll
        for (int v = 0; v < 8; ++v) {
            float t = fmaxf(st[0][v], st[1][v]);
            t = fmaxf(t, __shfl_xor(t, 1));
            t = fmaxf(t, __shfl_xor(t, 2));
            t = fmaxf(t, __shfl_xor(t, 4));
            t = fmaxf(t, __shfl_xor(t, 8));
            const float mn = fmaxf(m8[v], t);
            corr[v] = __expf(m8[v] - mn);
            m8[v] = mn;
            const float p0 = __expf(st[0][v] - mn);
            const float p1 = __expf(st[1][v] - mn);
            st[0][v] = p0;
            st[1][v] = p1;
            float s = p0 + p1;
            s += __shfl_xor(s, 1);
            s += __shfl_xor(s, 2);
            s += __shfl_xor(s, 4);
            s += __shfl_xor(s, 8);
            l8[v] = l8[v] * corr[v] + s;
        }
        #pragma unroll
        for (int t = 0; t < 8; ++t)
            #pragma unroll
            for (int v = 0; v < 8; ++v) zc[t][v] *= corr[v];

        // ---- P chunk to LDS as [i][j] ------------------------------------
        #pragma unroll
        for (int v = 0; v < 8; ++v) {
            sPw[(v + 8 * h) * PCH + nn]      = (__bf16)st[0][v];
            sPw[(v + 8 * h) * PCH + 16 + nn] = (__bf16)st[1][v];
        }
        // B operand = P: lane nn = row i, j runs 8h.. and 16+8h.. (2x ds_b128)
        const __bf16* ps = sPw + nn * PCH + 8 * h;
        v16bf bp = ld_2x128(ps, ps + 16);

        // ---- Z^T += V^T @ P : A from interleaved global V (2x b128/lane) --
        const __bf16* vbase = Vg + (jc * 4 + h) * 1024 + nn * 8;
        #pragma unroll
        for (int t = 0; t < 8; ++t) {
            const __bf16* va = vbase + t * 128;   // dim = t*16 + nn
            v16bf av = ld_2x128(va, va + 2048);   // j runs 8h.. and 16+8h..
            zc[t] = wmma_bf16(av, bp, zc[t]);
        }

        wait_async0();       // own async slice of next chunk complete
        __syncthreads();     // all waves' slices visible; cur reads finished
    }

    // ---- publish 1/rowsum per i (Z^T needs it per lane) --------------------
    if (nn == 0) {
        #pragma unroll
        for (int v = 0; v < 8; ++v) sSum[wid][v + 8 * h] = l8[v];
    }
    __syncthreads();
    const float rinvL = 1.0f / sSum[wid][nn];

    // ---- park z (normalized, un-transposed) in LDS; aliases Q buffers ------
    float* sZw = (float*)smem + wid * (16 * ZSTR);
    #pragma unroll
    for (int t = 0; t < 8; ++t)
        #pragma unroll
        for (int v = 0; v < 8; ++v)
            sZw[nn * ZSTR + t * 16 + v + 8 * h] = zc[t][v] * rinvL;
    __syncthreads();

    // ---- residual + LayerNorm (2 lanes per row, 64 dims each, vectorized) --
    const int r = nn;
    const int irow = i0 + r;
    const int node = s0 + min(irow, n - 1);
    const int d0 = h * 64;
    const float4* x4 = (const float4*)(x + (size_t)node * DD + d0);
    const float4* z4 = (const float4*)(sZw + r * ZSTR + d0);
    float4 hreg[16];
    float s1 = 0.f, s2 = 0.f;
    #pragma unroll
    for (int q = 0; q < 16; ++q) {
        float4 xv = x4[q];
        float4 zv = z4[q];
        float4 hv = make_float4(xv.x + zv.x, xv.y + zv.y, xv.z + zv.z, xv.w + zv.w);
        hreg[q] = hv;
        s1 += (hv.x + hv.y) + (hv.z + hv.w);
        s2 += (hv.x * hv.x + hv.y * hv.y) + (hv.z * hv.z + hv.w * hv.w);
    }
    s1 += __shfl_xor(s1, 16);
    s2 += __shfl_xor(s2, 16);
    const float mu = s1 * (1.f / 128.f);
    const float var = s2 * (1.f / 128.f) - mu * mu;
    const float rstd = rsqrtf(var + EPSV);
    if (irow < n) {
        float4* o4 = (float4*)(out + (size_t)node * DD + d0);
        const float4* g4 = (const float4*)(gamma + d0);
        const float4* b4 = (const float4*)(beta + d0);
        #pragma unroll
        for (int q = 0; q < 16; ++q) {
            float4 gv = g4[q];
            float4 bv = b4[q];
            float4 hv = hreg[q];
            o4[q] = make_float4((hv.x - mu) * rstd * gv.x + bv.x,
                                (hv.y - mu) * rstd * gv.y + bv.y,
                                (hv.z - mu) * rstd * gv.z + bv.z,
                                (hv.w - mu) * rstd * gv.w + bv.w);
        }
    }
}

// ---------------------------------------------------------------------------
// Host launcher
// ---------------------------------------------------------------------------
extern "C" void kernel_launch(void* const* d_in, const int* in_sizes, int n_in,
                              void* d_out, int out_size, void* d_ws, size_t ws_size,
                              hipStream_t stream) {
    const float* x     = (const float*)d_in[0];
    const float* Wk    = (const float*)d_in[1];
    const float* bk    = (const float*)d_in[2];
    const float* Wq    = (const float*)d_in[3];
    const float* bq    = (const float*)d_in[4];
    const float* Wv    = (const float*)d_in[5];
    const float* bv    = (const float*)d_in[6];
    const float* gamma = (const float*)d_in[7];
    const float* beta  = (const float*)d_in[8];
    const int*   batch = (const int*)d_in[9];
    float* out = (float*)d_out;

    char* ws = (char*)d_ws;
    const size_t ND = (size_t)NN * DD;
    __bf16* Kb  = (__bf16*)(void*)(ws);
    __bf16* Qb  = (__bf16*)(void*)(ws + ND * 2);
    __bf16* Vp8 = (__bf16*)(void*)(ws + 2 * ND * 2);
    int* counts = (int*)(void*)(ws + 2 * ND * 2 + (size_t)GG * PADG * 2);
    int* starts = counts + GG;

    // Zero V pad region (capture-legal) so pad reads are exact +0.0 bf16.
    hipMemsetAsync(Vp8, 0, (size_t)GG * PADG * 2, stream);

    zero_kernel<<<(GG + 255) / 256, 256, 0, stream>>>(counts);
    hist_kernel<<<NN / 256, 256, 0, stream>>>(batch, counts);
    scan_kernel<<<1, GG, 0, stream>>>(counts, starts);

    proj_kernel<<<dim3(NN / 32, 3), 32, 0, stream>>>(
        x, Wk, bk, Wq, bq, Wv, bv, batch, starts, Kb, Qb, Vp8);

    attn_kernel<<<dim3(LMAX / 64, GG), 128, 0, stream>>>(
        Kb, Qb, Vp8, x, gamma, beta, counts, starts, out);
}
